// DenseGCN_21869973471634
// MI455X (gfx1250) — matrix operand order repaired and checked by reading, not compiled, not run on hardware
//
#include <hip/hip_runtime.h>
#include <stdint.h>
#include <stddef.h>

// ---------------------------------------------------------------------------
// Types for CDNA5 WMMA
// ---------------------------------------------------------------------------
typedef __attribute__((ext_vector_type(16))) __bf16         v16bf;
typedef __attribute__((ext_vector_type(8)))  float          v8f;
typedef __attribute__((ext_vector_type(16))) unsigned short v16u;

static __device__ __forceinline__ unsigned short f2bf(float f) {
    // round-to-nearest-even float32 -> bf16
    unsigned int u = __float_as_uint(f);
    unsigned int r = u + 0x7FFFu + ((u >> 16) & 1u);
    return (unsigned short)(r >> 16);
}

// ---------------------------------------------------------------------------
// Multi-term GEMM:  C = act( sum_t  diag(scale_t) * A_t @ B_t  + bias )
// Each A_t is [N, K_t] f32 row-major, B_t is pre-packed bf16 in WMMA lane
// layout.  One wave computes one 16x16 tile of C via v_wmma_f32_16x16x32_bf16.
// ---------------------------------------------------------------------------
struct Term {
    const float*          A;     // [N, K] f32
    const unsigned short* Bp;    // packed bf16: [F/16][K/32][32 lanes][16]
    const float*          scale; // per-row scale (1/deg) or nullptr
    int                   K;     // multiple of 32
};
struct TermList {
    Term t[6];
    int  n;
};

__global__ __launch_bounds__(256)
void sage_gemm_wmma(TermList terms, const float* __restrict__ bias,
                    float* __restrict__ C, int F, int mTiles, int relu)
{
    const int lane  = threadIdx.x & 31;
    const int wave  = blockIdx.x * (blockDim.x >> 5) + (threadIdx.x >> 5);
    const int nTiles = F >> 4;
    const int mTile = wave / nTiles;
    const int nTile = wave - mTile * nTiles;
    if (mTile >= mTiles) return;            // wave-uniform: EXEC stays all-ones

    const int hf = lane >> 4;               // half-wave id (0/1)
    const int ln = lane & 15;
    const int m  = (mTile << 4) + ln;       // this lane's A row

    v8f c = {};

    for (int t = 0; t < terms.n; ++t) {
        const float*          A  = terms.t[t].A;
        const unsigned short* Bp = terms.t[t].Bp;
        const float*          sc = terms.t[t].scale;
        const int             K  = terms.t[t].K;
        const int        kchunks = K >> 5;

        const float s = sc ? sc[m] : 1.0f;
        const float* arow = A + (size_t)m * K;
        const v16u* bbase = (const v16u*)Bp + ((size_t)nTile * kchunks) * 32 + lane;

        for (int kc = 0; kc < kchunks; ++kc) {
            // ---- A fragment (16x32 bf16, ISA layout):
            // lanes 0-15 : K = kc*32 + [0..7], [16..23]
            // lanes 16-31: K = kc*32 + [8..15], [24..31]
            const float* ap = arow + (kc << 5) + (hf << 3);
            float4 f0 = *(const float4*)(ap);
            float4 f1 = *(const float4*)(ap + 4);
            float4 f2 = *(const float4*)(ap + 16);
            float4 f3 = *(const float4*)(ap + 20);
            float vals[16] = { f0.x, f0.y, f0.z, f0.w,  f1.x, f1.y, f1.z, f1.w,
                               f2.x, f2.y, f2.z, f2.w,  f3.x, f3.y, f3.z, f3.w };
            union { v16u u; v16bf b; } au;
#pragma unroll
            for (int i = 0; i < 16; ++i) au.u[i] = f2bf(vals[i] * s);

            // ---- B fragment: one contiguous 32-byte load per lane
            union { v16u u; v16bf b; } bu;
            bu.u = bbase[(size_t)kc * 32];

            c = __builtin_amdgcn_wmma_f32_16x16x32_bf16(
                    false, au.b, false, bu.b, (short)0, c, false, false);
        }
    }

    // ---- epilogue: bias + ReLU, C/D layout: VGPR r -> row r + 8*hf
    const float bv = bias[(nTile << 4) + ln];
    float* crow = C + (size_t)((mTile << 4) + (hf << 3)) * F + (nTile << 4) + ln;
#pragma unroll
    for (int r = 0; r < 8; ++r) {
        float v = c[r] + bv;
        if (relu) v = fmaxf(v, 0.0f);
        crow[(size_t)r * F] = v;
    }
}

// ---------------------------------------------------------------------------
// Pack a K-slice of W [rows r0..r0+Kc) of a [*,Ftot] f32 matrix] into the
// WMMA B-operand bf16 lane layout: out[((nt*Kc/32 + kc)*32 + lane)*16 + e]
//   column n = nt*16 + (lane&15),  row k = kc*32 + 8*(lane>>4) + (e&7) + 16*(e>>3)
// ---------------------------------------------------------------------------
__global__ void pack_weights(const float* __restrict__ W, int Ftot, int r0,
                             int Kc, unsigned short* __restrict__ out)
{
    int t = blockIdx.x * blockDim.x + threadIdx.x;
    int total = Kc * Ftot;
    if (t >= total) return;
    int e    = t & 15;
    int l    = (t >> 4) & 31;
    int rest = t >> 9;
    int kchunks = Kc >> 5;
    int kc = rest % kchunks;
    int nt = rest / kchunks;
    int n  = (nt << 4) + (l & 15);
    int k  = (kc << 5) + ((l >> 4) << 3) + (e & 7) + ((e >> 3) << 4);
    out[t] = f2bf(W[(size_t)(r0 + k) * Ftot + n]);
}

// ---------------------------------------------------------------------------
// Edge aggregation kernels (atomics resolve in the 192MB L2: agg arrays are
// 12.8-25.6 MB and stay resident)
// ---------------------------------------------------------------------------
__global__ void count_deg(const long long* __restrict__ dst,
                          float* __restrict__ cnt, long long E)
{
    long long t = (long long)blockIdx.x * blockDim.x + threadIdx.x;
    if (t >= E) return;
    atomicAdd(cnt + dst[t], 1.0f);
}

__global__ void recip_deg(float* __restrict__ cnt, int N)
{
    int t = blockIdx.x * blockDim.x + threadIdx.x;
    if (t < N) cnt[t] = 1.0f / fmaxf(cnt[t], 1.0f);
}

__global__ void scatter_add_f(const long long* __restrict__ src,
                              const long long* __restrict__ dst,
                              const float* __restrict__ X,
                              float* __restrict__ AGG,
                              long long E, int F)
{
    const int c4 = F >> 2;                       // float4 chunks per row
    long long total = E * c4;
    long long t = (long long)blockIdx.x * blockDim.x + threadIdx.x;
    if (t >= total) return;
    long long e = t / c4;
    int c = (int)(t - e * c4);
    long long s = src[e], d = dst[e];
    const float4 v = ((const float4*)(X + s * F))[c];
    float* o = AGG + d * F + ((long long)c << 2);
    atomicAdd(o + 0, v.x);
    atomicAdd(o + 1, v.y);
    atomicAdd(o + 2, v.z);
    atomicAdd(o + 3, v.w);
}

// ---------------------------------------------------------------------------
// Host-side orchestration
// ---------------------------------------------------------------------------
extern "C" void kernel_launch(void* const* d_in, const int* in_sizes, int n_in,
                              void* d_out, int out_size, void* d_ws, size_t ws_size,
                              hipStream_t stream) {
    (void)n_in; (void)out_size; (void)ws_size;

    const float*     x  = (const float*)d_in[0];
    const long long* ei = (const long long*)d_in[1];       // int64 edge_index [2,E]
    const float* Wp  = (const float*)d_in[2];  const float* bp  = (const float*)d_in[3];
    const float* Wl1 = (const float*)d_in[4];  const float* bl1 = (const float*)d_in[5];
    const float* Wr1 = (const float*)d_in[6];
    const float* Wl2 = (const float*)d_in[7];  const float* bl2 = (const float*)d_in[8];
    const float* Wr2 = (const float*)d_in[9];
    const float* Wl3 = (const float*)d_in[10]; const float* bl3 = (const float*)d_in[11];
    const float* Wr3 = (const float*)d_in[12];

    const int       N = in_sizes[0] / 64;                  // 50000
    const long long E = in_sizes[1] / 2;                   // 800000
    const long long* srcp = ei;
    const long long* dstp = ei + E;
    const int mTiles = N / 16;                             // 3125

    // ---- carve workspace (256B aligned slices) ----
    char* wp = (char*)d_ws;
    auto carve = [&](size_t bytes) -> void* {
        void* r = (void*)wp;
        wp += (bytes + 255) & ~(size_t)255;
        return r;
    };
    float* degi   = (float*)carve((size_t)N * 4);
    float* agg_x  = (float*)carve((size_t)N * 64 * 4);
    float* x_p    = (float*)carve((size_t)N * 64 * 4);
    float* h1     = (float*)carve((size_t)N * 64 * 4);
    float* agg_xp = (float*)carve((size_t)N * 64 * 4);
    float* agg_h1 = (float*)carve((size_t)N * 64 * 4);
    float* h2     = (float*)carve((size_t)N * 128 * 4);
    float* agg_h2 = (float*)carve((size_t)N * 128 * 4);
    unsigned short* Pp   = (unsigned short*)carve(64 * 64 * 2);
    unsigned short* Pl1  = (unsigned short*)carve(64 * 64 * 2);
    unsigned short* Pr1  = (unsigned short*)carve(64 * 64 * 2);
    unsigned short* Pl2a = (unsigned short*)carve(64 * 128 * 2);
    unsigned short* Pl2b = (unsigned short*)carve(64 * 128 * 2);
    unsigned short* Pr2a = (unsigned short*)carve(64 * 128 * 2);
    unsigned short* Pr2b = (unsigned short*)carve(64 * 128 * 2);
    unsigned short* Pl3a = (unsigned short*)carve(64 * 128 * 2);
    unsigned short* Pl3b = (unsigned short*)carve(64 * 128 * 2);
    unsigned short* Pl3c = (unsigned short*)carve(128 * 128 * 2);
    unsigned short* Pr3a = (unsigned short*)carve(64 * 128 * 2);
    unsigned short* Pr3b = (unsigned short*)carve(64 * 128 * 2);
    unsigned short* Pr3c = (unsigned short*)carve(128 * 128 * 2);

    // ---- zero accumulators ----
    hipMemsetAsync(degi,   0, (size_t)N * 4,        stream);
    hipMemsetAsync(agg_x,  0, (size_t)N * 64 * 4,   stream);
    hipMemsetAsync(agg_xp, 0, (size_t)N * 64 * 4,   stream);
    hipMemsetAsync(agg_h1, 0, (size_t)N * 64 * 4,   stream);
    hipMemsetAsync(agg_h2, 0, (size_t)N * 128 * 4,  stream);

    // ---- pack weights into WMMA bf16 layout ----
    auto pack = [&](const float* W, int Ftot, int r0, int Kc, unsigned short* out) {
        int total = Kc * Ftot;
        pack_weights<<<(total + 255) / 256, 256, 0, stream>>>(W, Ftot, r0, Kc, out);
    };
    pack(Wp,  64,   0,  64, Pp);
    pack(Wl1, 64,   0,  64, Pl1);
    pack(Wr1, 64,   0,  64, Pr1);
    pack(Wl2, 128,  0,  64, Pl2a);  pack(Wl2, 128,  64,  64, Pl2b);
    pack(Wr2, 128,  0,  64, Pr2a);  pack(Wr2, 128,  64,  64, Pr2b);
    pack(Wl3, 128,  0,  64, Pl3a);  pack(Wl3, 128,  64,  64, Pl3b);
    pack(Wl3, 128, 128, 128, Pl3c);
    pack(Wr3, 128,  0,  64, Pr3a);  pack(Wr3, 128,  64,  64, Pr3b);
    pack(Wr3, 128, 128, 128, Pr3c);

    // ---- degree + first aggregation (agg_x = segsum x[src] by dst) ----
    count_deg<<<(int)((E + 255) / 256), 256, 0, stream>>>(dstp, degi, E);
    recip_deg<<<(N + 255) / 256, 256, 0, stream>>>(degi, N);
    {
        long long total = E * 16;  // F=64 -> 16 float4 per edge
        scatter_add_f<<<(int)((total + 255) / 256), 256, 0, stream>>>(
            srcp, dstp, x, agg_x, E, 64);
    }

    auto gemm = [&](TermList tl, const float* bias, float* C, int F) {
        int waves  = mTiles * (F >> 4);
        int blocks = (waves + 7) / 8;      // 8 waves per 256-thread block
        sage_gemm_wmma<<<blocks, 256, 0, stream>>>(tl, bias, C, F, mTiles, 1);
    };

    // ---- x_p = relu(x @ Wp + bp) ----
    {
        TermList tl; tl.n = 1;
        tl.t[0] = { x, Pp, nullptr, 64 };
        gemm(tl, bp, x_p, 64);
    }

    // ---- h1 = relu( mean(x) @ Wl1 + x @ Wr1 + bl1 ) ----
    {
        TermList tl; tl.n = 2;
        tl.t[0] = { agg_x, Pl1, degi,    64 };
        tl.t[1] = { x,     Pr1, nullptr, 64 };
        gemm(tl, bl1, h1, 64);
    }

    // ---- aggregate x_p and h1 (reused by BOTH layer 2 and layer 3) ----
    {
        long long total = E * 16;
        scatter_add_f<<<(int)((total + 255) / 256), 256, 0, stream>>>(
            srcp, dstp, x_p, agg_xp, E, 64);
        scatter_add_f<<<(int)((total + 255) / 256), 256, 0, stream>>>(
            srcp, dstp, h1, agg_h1, E, 64);
    }

    // ---- h2 = relu( mean([x_p,h1]) @ Wl2 + [x_p,h1] @ Wr2 + bl2 ) ----
    {
        TermList tl; tl.n = 4;
        tl.t[0] = { agg_xp, Pl2a, degi,    64 };
        tl.t[1] = { agg_h1, Pl2b, degi,    64 };
        tl.t[2] = { x_p,    Pr2a, nullptr, 64 };
        tl.t[3] = { h1,     Pr2b, nullptr, 64 };
        gemm(tl, bl2, h2, 128);
    }

    // ---- aggregate h2 only (agg_xp/agg_h1 reused) ----
    {
        long long total = E * 32;  // F=128 -> 32 float4 per edge
        scatter_add_f<<<(int)((total + 255) / 256), 256, 0, stream>>>(
            srcp, dstp, h2, agg_h2, E, 128);
    }

    // ---- h3 = relu( mean([x_p,h1,h2]) @ Wl3 + [x_p,h1,h2] @ Wr3 + bl3 ) ----
    {
        TermList tl; tl.n = 6;
        tl.t[0] = { agg_xp, Pl3a, degi,    64  };
        tl.t[1] = { agg_h1, Pl3b, degi,    64  };
        tl.t[2] = { agg_h2, Pl3c, degi,    128 };
        tl.t[3] = { x_p,    Pr3a, nullptr, 64  };
        tl.t[4] = { h1,     Pr3b, nullptr, 64  };
        tl.t[5] = { h2,     Pr3c, nullptr, 128 };
        gemm(tl, bl3, (float*)d_out, 128);
    }
}
